// SplitOneEncoder_41532333752863
// MI455X (gfx1250) — matrix operand order repaired and checked by reading, not compiled, not run on hardware
//
#include <hip/hip_runtime.h>

// ---------------------------------------------------------------------------
// Problem constants
// ---------------------------------------------------------------------------
#define BATCH 32
#define LEN   1024
#define DIM   1024
#define NHEAD 8
#define EDIM  128
#define NMODE 64
#define F2    128          // 2*NMODE (Re rows then Im rows)
#define KWIN  25
#define PADW  12

typedef __attribute__((ext_vector_type(16))) __bf16    bf16x16;
typedef __attribute__((ext_vector_type(8)))  float     floatx8;
typedef __attribute__((ext_vector_type(4)))  unsigned  uint32x4;
typedef __attribute__((ext_vector_type(8)))  int       int32x8;
typedef __attribute__((ext_vector_type(4)))  int       int32x4;

// fp32 -> bf16 RNE via integer path
__device__ __forceinline__ __bf16 f2bf(float f) {
    union { float f; unsigned u; } a; a.f = f;
    unsigned r = (a.u + 0x7FFFu + ((a.u >> 16) & 1u)) >> 16;
    union { unsigned short s; __bf16 b; } o; o.s = (unsigned short)r;
    return o.b;
}

// ---------------------------------------------------------------------------
// WMMA fragment loaders (wave32, 16x16x32 bf16). LDS tiles have pitch 40 bf16
// (80B rows = 64B data + 16B pad, produced natively by the TDM pad engine).
// A frag: lane%16 = M row; K halves 0-15 / 16-31
// B frag (tile stored [n][k]): lane%16 = N col; K halves 0-15 / 16-31
// ---------------------------------------------------------------------------
__device__ __forceinline__ bf16x16 load_fragA(const __bf16* tile, int mrow, int lane) {
    const __bf16* row = tile + mrow * 40;
    const int kb = (lane & 16) ? 16 : 0;
    union { bf16x16 v; unsigned u[8]; } t;
    const unsigned* p0 = (const unsigned*)(row + kb);
    const unsigned* p1 = (const unsigned*)(row + kb + 8);
#pragma unroll
    for (int q = 0; q < 4; ++q) { t.u[q] = p0[q]; t.u[4 + q] = p1[q]; }
    return t.v;
}

__device__ __forceinline__ bf16x16 load_fragB(const __bf16* tileT, int ncol, int lane) {
    const __bf16* row = tileT + ncol * 40 + ((lane & 16) ? 16 : 0);
    union { bf16x16 v; unsigned u[8]; } t;
    const unsigned* p = (const unsigned*)row;
#pragma unroll
    for (int q = 0; q < 8; ++q) t.u[q] = p[q];
    return t.v;
}

__device__ __forceinline__ bf16x16 neg_frag(bf16x16 a) {
    union { bf16x16 v; unsigned u[8]; } t; t.v = a;
#pragma unroll
    for (int q = 0; q < 8; ++q) t.u[q] ^= 0x80008000u;
    return t.v;
}

#define WMMA_BF16(acc, a, b) \
    acc = __builtin_amdgcn_wmma_f32_16x16x32_bf16(false, (a), false, (b), (short)0, (acc), false, false)

// ---------------------------------------------------------------------------
// Tensor Data Mover tile load: 128 rows x 32 bf16 from a row-major
// [tensor_d1 x tensor_d0] bf16 tensor into LDS, 16B pad per 64B row
// (-> LDS pitch 80B = 40 bf16). Descriptor per CDNA5 ISA ch. 8.
// ---------------------------------------------------------------------------
#if __has_builtin(__builtin_amdgcn_tensor_load_to_lds)
#define USE_TDM 1
#else
#define USE_TDM 0
#endif

#if USE_TDM
__device__ __forceinline__ void tdm_load_tile128x32(
    const __bf16* gsrc, unsigned lds_byte_off,
    unsigned tensor_d0, unsigned tensor_d1, unsigned row_stride_elems)
{
    unsigned long long ga = (unsigned long long)(size_t)gsrc;
    uint32x4 g0;
    g0[0] = 1u;                                          // count=1, user descriptor
    g0[1] = lds_byte_off;                                // lds_addr
    g0[2] = (unsigned)(ga & 0xFFFFFFFFu);                // global_addr[31:0]
    g0[3] = (unsigned)((ga >> 32) & 0x01FFFFFFu) | 0x80000000u; // addr[56:32] | type=2

    int32x8 g1;
    // data_size=2B (1), pad_enable, pad_interval=16DW (3), pad_amount=4DW (3)
    g1[0] = (1 << 16) | (1 << 20) | (3 << 22) | (3 << 25);
    g1[1] = (int)((tensor_d0 & 0xFFFFu) << 16);                          // dim0[15:0]
    g1[2] = (int)((tensor_d0 >> 16) | ((tensor_d1 & 0xFFFFu) << 16));    // dim0[31:16]|dim1[15:0]
    g1[3] = (int)((tensor_d1 >> 16) | (32u << 16));                      // dim1[31:16]|tile_dim0=32
    g1[4] = 128;                                                         // tile_dim1=128, tile_dim2=0
    g1[5] = (int)row_stride_elems;                                       // dim0_stride[31:0]
    g1[6] = 0;
    g1[7] = 0;

    int32x4 g2; g2[0] = 1; g2[1] = 0; g2[2] = 0; g2[3] = 0;  // 2D (tensor_dim2=1)
    int32x4 g3; g3[0] = 0; g3[1] = 0; g3[2] = 0; g3[3] = 0;

#if __has_include(<hip/amd_detail/amd_gfx1250_TDM.h>)
    int32x8 g4; g4[0]=0; g4[1]=0; g4[2]=0; g4[3]=0; g4[4]=0; g4[5]=0; g4[6]=0; g4[7]=0;
    __builtin_amdgcn_tensor_load_to_lds(g0, g1, g2, g3, g4, 0);
#else
    __builtin_amdgcn_tensor_load_to_lds(g0, g1, g2, g3, 0);
#endif
}
#endif

// Fallback staging when TDM builtin is unavailable: bf16 dword copies
__device__ __forceinline__ void stage_tile_manual(
    const __bf16* __restrict__ g, int row_stride_elems, __bf16* lds, int tid)
{
#pragma unroll
    for (int i = 0; i < 8; ++i) {
        int idx = tid + i * 256;        // 2048 dwords = 128 rows x 16 dwords
        int r = idx >> 4, c = idx & 15;
        ((unsigned*)lds)[r * 20 + c] =
            ((const unsigned*)(g + (long long)r * row_stride_elems))[c];
    }
}

// ---------------------------------------------------------------------------
// Packing kernels (one-shot fp32 -> bf16; weights transposed so every GEMM
// operand is row-major over K and thus TDM-tileable)
// ---------------------------------------------------------------------------
__global__ __launch_bounds__(256) void pack_bf16_kernel(
    const float* __restrict__ src, __bf16* __restrict__ dst, int n)
{
    int t = blockIdx.x * 256 + threadIdx.x;
    if (t < n) dst[t] = f2bf(src[t]);
}

__global__ __launch_bounds__(256) void transpose_pack_kernel(
    const float* __restrict__ W, __bf16* __restrict__ WT, int R, int C)
{
    int t = blockIdx.x * 256 + threadIdx.x;
    if (t >= R * C) return;
    int r = t / C, c = t - r * C;             // coalesced read along c
    WT[(long long)c * R + r] = f2bf(W[t]);
}

// basisFb[m'][l] (A of DFT GEMM) and basisIT[l][m'] (Bt of iDFT GEMM)
__global__ __launch_bounds__(256) void init_basis_kernel(
    __bf16* __restrict__ bF, __bf16* __restrict__ bIT)
{
    int t = blockIdx.x * 256 + threadIdx.x;
    if (t >= F2 * LEN) return;
    int mp = t >> 10, l = t & (LEN - 1);
    int m = mp & (NMODE - 1);
    float ang = 6.283185307179586f * (float)m * (float)l * (1.0f / (float)LEN);
    float v = (mp < NMODE) ? __cosf(ang) : -__sinf(ang);
    bF[t] = f2bf(v);
    float scale = ((m == 0) ? 1.0f : 2.0f) * (1.0f / (float)LEN);
    bIT[(long long)l * F2 + mp] = f2bf(v * scale);
}

// ---------------------------------------------------------------------------
// Generic bf16 WMMA GEMM:  C = A[M,K] @ Bt[N,K]^T  (+bias) (+resid)
// Double-buffered TDM pipeline: wave0 streams A tiles, wave1 streams B tiles;
// prefetch for k+32 is issued before computing k, so DMA overlaps the WMMAs.
// OMODE 0: C row-major [M,N] (+ batch stride sC)
// OMODE 1: gm=(b,l) split -> C[b][n][l]  (per-batch transposed, for q^T)
// OMODE 2: C[z][n][m]                    (per-batch transposed, for Y[B,L,D])
// ---------------------------------------------------------------------------
template <bool HAS_BIAS, bool HAS_RESID, bool OUT_BF16, int OMODE>
__global__ __launch_bounds__(256) void gemm_bf16_kernel(
    const __bf16* __restrict__ A, const __bf16* __restrict__ Bt,
    const float* __restrict__ bias, const float* __restrict__ resid,
    void* __restrict__ Cout, int M, int N, int K,
    long long sA, long long sB, long long sC)
{
    __shared__ __bf16 ldsA[2][128 * 40];
    __shared__ __bf16 ldsB[2][128 * 40];

    const int tid  = threadIdx.x;
    const int lane = tid & 31;
    const int wave = tid >> 5;
    const int wm   = wave & 1;
    const int wn   = wave >> 1;
    const int bm   = blockIdx.y * 128;
    const int bn   = blockIdx.x * 128;
    const int z    = blockIdx.z;

    A  += (long long)z * sA;
    Bt += (long long)z * sB;

    floatx8 acc[4][2];
#pragma unroll
    for (int i = 0; i < 4; ++i)
#pragma unroll
        for (int j = 0; j < 2; ++j) acc[i][j] = (floatx8)0.0f;

#if USE_TDM
    const unsigned offA0 = (unsigned)(size_t)(void*)&ldsA[0][0];
    const unsigned offA1 = (unsigned)(size_t)(void*)&ldsA[1][0];
    const unsigned offB0 = (unsigned)(size_t)(void*)&ldsB[0][0];
    const unsigned offB1 = (unsigned)(size_t)(void*)&ldsB[1][0];

    // prologue: fill buffer 0
    if (wave == 0)      tdm_load_tile128x32(A  + (long long)bm * K, offA0,
                                            (unsigned)K, (unsigned)M, (unsigned)K);
    else if (wave == 1) tdm_load_tile128x32(Bt + (long long)bn * K, offB0,
                                            (unsigned)K, (unsigned)N, (unsigned)K);
#endif

    int cur = 0;
    for (int k0 = 0; k0 < K; k0 += 32, cur ^= 1) {
#if USE_TDM
        const bool has_next = (k0 + 32) < K;
        if (wave == 0) {
            if (has_next) {
                tdm_load_tile128x32(A + (long long)bm * K + (k0 + 32),
                                    cur ? offA0 : offA1,
                                    (unsigned)K, (unsigned)M, (unsigned)K);
                // TDM ops complete in order per wave: <=1 outstanding means the
                // current buffer's load is done while the prefetch is in flight
                __builtin_amdgcn_s_wait_tensorcnt(1);
            } else {
                __builtin_amdgcn_s_wait_tensorcnt(0);
            }
        } else if (wave == 1) {
            if (has_next) {
                tdm_load_tile128x32(Bt + (long long)bn * K + (k0 + 32),
                                    cur ? offB0 : offB1,
                                    (unsigned)K, (unsigned)N, (unsigned)K);
                __builtin_amdgcn_s_wait_tensorcnt(1);
            } else {
                __builtin_amdgcn_s_wait_tensorcnt(0);
            }
        }
        __syncthreads();   // publish buffer `cur` to all waves
#else
        __syncthreads();
        stage_tile_manual(A  + (long long)bm * K + k0, K, &ldsA[0][0], tid);
        stage_tile_manual(Bt + (long long)bn * K + k0, K, &ldsB[0][0], tid);
        __syncthreads();
        cur = 0;
#endif
        const __bf16* tA = &ldsA[cur][0];
        const __bf16* tB = &ldsB[cur][0];

        bf16x16 bfr[2];
#pragma unroll
        for (int jn = 0; jn < 2; ++jn)
            bfr[jn] = load_fragB(tB, wn * 32 + jn * 16 + (lane & 15), lane);
#pragma unroll
        for (int im = 0; im < 4; ++im) {
            bf16x16 afr = load_fragA(tA, wm * 64 + im * 16 + (lane & 15), lane);
#pragma unroll
            for (int jn = 0; jn < 2; ++jn) WMMA_BF16(acc[im][jn], afr, bfr[jn]);
        }
#if USE_TDM
        __syncthreads();   // all reads done before next iteration's DMA overwrites
#endif
    }

    const int mb = (lane & 16) ? 8 : 0;
    const int cn = lane & 15;
#pragma unroll
    for (int im = 0; im < 4; ++im)
#pragma unroll
        for (int jn = 0; jn < 2; ++jn) {
            const int gn = bn + wn * 32 + jn * 16 + cn;
            float badd = 0.0f;
            if constexpr (HAS_BIAS) badd = bias[gn];
#pragma unroll
            for (int r = 0; r < 8; ++r) {
                const int gm = bm + wm * 64 + im * 16 + mb + r;
                float v = acc[im][jn][r] + badd;
                long long off;
                if constexpr (OMODE == 0) {
                    off = (long long)z * sC + (long long)gm * N + gn;
                } else if constexpr (OMODE == 1) {
                    const int bb = gm >> 10, ll = gm & (LEN - 1);
                    off = (long long)bb * (DIM * LEN) + (long long)gn * LEN + ll;
                } else {
                    off = (long long)z * sC + (long long)gn * DIM + gm;
                }
                if constexpr (HAS_RESID) v += resid[off];
                if constexpr (OUT_BF16) ((__bf16*)Cout)[off] = f2bf(v);
                else                    ((float*)Cout)[off]  = v;
            }
        }
}

// ---------------------------------------------------------------------------
// Mode mixing: one block per (h, m). Complex GEMM 32x128x128 as 4 real WMMAs.
//   Xft layout    [b][m' Re0..63 / Im64..127][d = h*128+e]   (bf16)
//   OutSelT layout [b][d = h*128+o][m']                      (bf16, A of iDFT)
// ---------------------------------------------------------------------------
__global__ __launch_bounds__(256) void modemix_kernel(
    const __bf16* __restrict__ Xft, const float* __restrict__ w1r,
    const float* __restrict__ w1i, __bf16* __restrict__ OutSelT)
{
    __shared__ __bf16 Xr[32 * 40], Xi[32 * 40];
    __shared__ __bf16 Wrt[128 * 40], Wit[128 * 40];   // [o][e]

    const int tid  = threadIdx.x;
    const int lane = tid & 31;
    const int wave = tid >> 5;
    const int h = blockIdx.x >> 6;
    const int m = blockIdx.x & 63;

    floatx8 accRe[2], accIm[2];
#pragma unroll
    for (int t2 = 0; t2 < 2; ++t2) { accRe[t2] = (floatx8)0.0f; accIm[t2] = (floatx8)0.0f; }

    for (int e0 = 0; e0 < EDIM; e0 += 32) {
        // X slices: bf16 -> LDS dword copies (no conversion)
#pragma unroll
        for (int i = 0; i < 2; ++i) {
            int idx = tid + i * 256;          // 512 dwords = 32 rows x 16 dwords
            int bb = idx >> 4, ep = idx & 15;
            long long rowR = (long long)bb * (F2 * DIM) + (long long)m * DIM + h * EDIM + e0;
            long long rowI = rowR + (long long)NMODE * DIM;
            ((unsigned*)Xr)[bb * 20 + ep] = ((const unsigned*)(Xft + rowR))[ep];
            ((unsigned*)Xi)[bb * 20 + ep] = ((const unsigned*)(Xft + rowI))[ep];
        }
        // weight slices: strided fp32 gather, convert, transpose into LDS
#pragma unroll
        for (int i = 0; i < 16; ++i) {
            int idx = tid + i * 256;
            int ee = idx >> 7, oo = idx & 127;
            long long widx = ((long long)(h * EDIM + e0 + ee) * EDIM + oo) * NMODE + m;
            Wrt[oo * 40 + ee] = f2bf(w1r[widx]);
            Wit[oo * 40 + ee] = f2bf(w1i[widx]);
        }
        __syncthreads();

        bf16x16 br = load_fragB(Wrt, wave * 16 + (lane & 15), lane);
        bf16x16 bi = load_fragB(Wit, wave * 16 + (lane & 15), lane);
#pragma unroll
        for (int t2 = 0; t2 < 2; ++t2) {
            bf16x16 ar  = load_fragA(Xr, t2 * 16 + (lane & 15), lane);
            bf16x16 ai  = load_fragA(Xi, t2 * 16 + (lane & 15), lane);
            bf16x16 ain = neg_frag(ai);
            WMMA_BF16(accRe[t2], ar,  br);
            WMMA_BF16(accRe[t2], ain, bi);
            WMMA_BF16(accIm[t2], ar,  bi);
            WMMA_BF16(accIm[t2], ai,  br);
        }
        __syncthreads();
    }

    const int mb = (lane & 16) ? 8 : 0;
    const int o  = wave * 16 + (lane & 15);
#pragma unroll
    for (int t2 = 0; t2 < 2; ++t2)
#pragma unroll
        for (int r = 0; r < 8; ++r) {
            int bb = t2 * 16 + mb + r;
            long long base = (long long)bb * (DIM * F2) + (long long)(h * EDIM + o) * F2;
            OutSelT[base + m]         = f2bf(accRe[t2][r]);
            OutSelT[base + NMODE + m] = f2bf(accIm[t2][r]);
        }
}

// ---------------------------------------------------------------------------
// series_decomp in place on x2 (= d_out): res = x2 - movmean_25(replicate pad)
// ---------------------------------------------------------------------------
__global__ __launch_bounds__(256) void decomp_kernel(float* __restrict__ X) {
    int t = blockIdx.x * blockDim.x + threadIdx.x;   // B*D threads
    int b = t >> 10, d = t & (DIM - 1);
    float* col = X + (long long)b * (LEN * DIM) + d;

    float ring[KWIN];
    float sum = 0.0f;
#pragma unroll
    for (int j = 0; j < KWIN; ++j) {
        int idx = j - PADW; idx = idx < 0 ? 0 : idx;
        float v = col[idx * DIM];
        ring[j] = v; sum += v;
    }
    for (int l = 0; l < LEN; ++l) {
        col[l * DIM] = ring[PADW] - sum * (1.0f / (float)KWIN);
        int nx = l + PADW + 1; nx = nx > (LEN - 1) ? (LEN - 1) : nx;
        float nv = col[nx * DIM];
        sum += nv - ring[0];
#pragma unroll
        for (int j = 0; j < KWIN - 1; ++j) ring[j] = ring[j + 1];
        ring[KWIN - 1] = nv;
    }
}

// ---------------------------------------------------------------------------
extern "C" void kernel_launch(void* const* d_in, const int* in_sizes, int n_in,
                              void* d_out, int out_size, void* d_ws, size_t ws_size,
                              hipStream_t stream) {
    const float* x   = (const float*)d_in[0];
    const float* Wq  = (const float*)d_in[1];
    const float* bq  = (const float*)d_in[2];
    const float* w1r = (const float*)d_in[3];
    const float* w1i = (const float*)d_in[4];
    const float* Wo  = (const float*)d_in[5];
    const float* bo  = (const float*)d_in[6];
    float* out = (float*)d_out;

    // workspace carve-up (all bf16)
    __bf16* p = (__bf16*)d_ws;
    __bf16* xb      = p; p += (long long)BATCH * LEN * DIM;   // x as bf16
    __bf16* WqT     = p; p += (long long)DIM * DIM;           // Wq^T
    __bf16* WoT     = p; p += (long long)DIM * DIM;           // Wo^T
    __bf16* basisFb = p; p += F2 * LEN;                       // DFT rows
    __bf16* basisIT = p; p += LEN * F2;                       // iDFT basis^T
    __bf16* q_t     = p; p += (long long)BATCH * DIM * LEN;   // q^T per batch [D,L]
    __bf16* Xft     = p; p += (long long)BATCH * F2 * DIM;    // [b][m'][d]
    __bf16* OutSelT = p; p += (long long)BATCH * DIM * F2;    // [b][d][m']
    __bf16* Y       = p; p += (long long)BATCH * LEN * DIM;   // [B,L,D]

    const int NBL = BATCH * LEN;

    // one-shot packing
    pack_bf16_kernel<<<NBL * DIM / 256, 256, 0, stream>>>(x, xb, NBL * DIM);
    transpose_pack_kernel<<<DIM * DIM / 256, 256, 0, stream>>>(Wq, WqT, DIM, DIM);
    transpose_pack_kernel<<<DIM * DIM / 256, 256, 0, stream>>>(Wo, WoT, DIM, DIM);
    init_basis_kernel<<<F2 * LEN / 256, 256, 0, stream>>>(basisFb, basisIT);

    // q^T = (x @ Wq + bq)^T per batch     [OMODE1, bf16 out]
    dim3 gBig(DIM / 128, NBL / 128, 1);
    gemm_bf16_kernel<true, false, true, 1><<<gBig, 256, 0, stream>>>(
        xb, WqT, bq, nullptr, q_t, NBL, DIM, DIM, 0, 0, 0);

    // truncated rfft: Xft_b = basisF @ q_b  (Bt = q^T per batch)
    dim3 gDft(DIM / 128, 1, BATCH);
    gemm_bf16_kernel<false, false, true, 0><<<gDft, 256, 0, stream>>>(
        basisFb, q_t, nullptr, nullptr, Xft, F2, DIM, LEN,
        0, (long long)DIM * LEN, (long long)F2 * DIM);

    // complex mode mixing -> OutSel^T
    modemix_kernel<<<NHEAD * NMODE, 256, 0, stream>>>(Xft, w1r, w1i, OutSelT);

    // truncated irfft: Y[b,l,d] = OutSelT[b] @ basisI  (transposed write, OMODE2)
    dim3 gIdft(LEN / 128, DIM / 128, BATCH);
    gemm_bf16_kernel<false, false, true, 2><<<gIdft, 256, 0, stream>>>(
        OutSelT, basisIT, nullptr, nullptr, Y, DIM, LEN, F2,
        (long long)DIM * F2, 0, (long long)LEN * DIM);

    // x2 = Y @ Wo + bo + x -> d_out (fp32)
    gemm_bf16_kernel<true, true, false, 0><<<gBig, 256, 0, stream>>>(
        Y, WoT, bo, x, out, NBL, DIM, DIM, 0, 0, 0);

    // res = x2 - movmean(x2), in place
    decomp_kernel<<<(BATCH * DIM) / 256, 256, 0, stream>>>(out);
}
